// DSMoE_84585085927449
// MI455X (gfx1250) — matrix operand order repaired and checked by reading, not compiled, not run on hardware
//
#include <hip/hip_runtime.h>

#define E_EXP 32
#define CDIM  256
#define HDIM  1024
#define MTILE 32      // token rows per block = 2 WMMA M-tiles

typedef __attribute__((ext_vector_type(16))) __bf16 v16bf;
typedef __attribute__((ext_vector_type(8)))  float  v8f;

struct U8x32 { uint4 a, b; };   // 32 bytes -> bit_cast to v16bf

// ---------------------------------------------------------------------------
// Kernel 1: gating. One wave32 per token. Lane e owns expert e's logit.
// ---------------------------------------------------------------------------
__global__ __launch_bounds__(256)
void moe_gate(const float* __restrict__ x, const float* __restrict__ Wg,
              float* __restrict__ ws_out, int* __restrict__ top_i,
              float* __restrict__ top_w, int n) {
    const int wave  = threadIdx.x >> 5;
    const int lane  = threadIdx.x & 31;
    const int token = blockIdx.x * (blockDim.x >> 5) + wave;
    if (token >= n) return;

    const float* xr = x + (size_t)token * CDIM;   // wave-uniform -> scalar loads
    float acc = 0.f;
    #pragma unroll 4
    for (int i = 0; i < CDIM; ++i)
        acc += xr[i] * Wg[i * E_EXP + lane];      // lane-coalesced gate weights

    // softmax across 32 lanes
    float m = acc;
    for (int off = 16; off > 0; off >>= 1) m = fmaxf(m, __shfl_xor(m, off));
    float p = __expf(acc - m);
    float s = p;
    for (int off = 16; off > 0; off >>= 1) s += __shfl_xor(s, off);
    float prob = p / s;

    float m1 = prob;
    for (int off = 16; off > 0; off >>= 1) m1 = fmaxf(m1, __shfl_xor(m1, off));
    unsigned long long b1 = __ballot(prob == m1);
    int a1 = __ffsll(b1) - 1;

    float pm = (lane == a1) ? -1.0f : prob;
    float m2 = pm;
    for (int off = 16; off > 0; off >>= 1) m2 = fmaxf(m2, __shfl_xor(m2, off));
    unsigned long long b2 = __ballot(pm == m2);
    int a2 = __ffsll(b2) - 1;

    float denom = fmaxf(m1 + m2, 1e-6f);
    float w1 = m1 / denom, w2 = m2 / denom;

    ws_out[(size_t)token * E_EXP + lane] =
        (lane == a1) ? w1 : ((lane == a2) ? w2 : 0.f);

    if (lane == 0) {
        top_i[token * 2 + 0] = a1;  top_i[token * 2 + 1] = a2;
        top_w[token * 2 + 0] = w1;  top_w[token * 2 + 1] = w2;
    }
}

// ---------------------------------------------------------------------------
// Kernel 2: deterministic per-expert token-list compaction (one wave/expert).
// ---------------------------------------------------------------------------
__global__ __launch_bounds__(32)
void moe_compact(const int* __restrict__ top_i, const float* __restrict__ top_w,
                 int* __restrict__ counts, int* __restrict__ tok_list,
                 float* __restrict__ wgt_list, int n) {
    const int e = blockIdx.x;
    const int lane = threadIdx.x;
    int base = 0;
    for (int t0 = 0; t0 < n; t0 += 32) {
        int t = t0 + lane;
        int which = -1;
        if (t < n) {
            if      (top_i[t * 2 + 0] == e) which = 0;
            else if (top_i[t * 2 + 1] == e) which = 1;
        }
        unsigned long long mask = __ballot(which >= 0);
        if (which >= 0) {
            int pre = __popcll(mask & ((1ull << lane) - 1ull));
            tok_list[(size_t)e * n + base + pre] = t;
            wgt_list[(size_t)e * n + base + pre] = top_w[t * 2 + which];
        }
        base += __popcll(mask);
    }
    if (lane == 0) counts[e] = base;
}

// ---------------------------------------------------------------------------
// Kernel 3: grouped-GEMM expert MLP via bf16 WMMA.
// Block = 32-token tile (2 WMMA M-tiles) of one expert, 256 threads (8 wave32).
// Each B fragment feeds 2 WMMAs. GEMM1 runs as two 4-N-tile passes so the
// accumulator footprint stays at 64 VGPRs and B loads pipeline deeply.
// x tile staged f32 via async-to-LDS DMA; staging buffer aliases actA.
// ---------------------------------------------------------------------------
__global__ __launch_bounds__(256, 2)
void moe_mlp(const float* __restrict__ x, const float* __restrict__ W1,
             const float* __restrict__ W2, const int* __restrict__ counts,
             const int* __restrict__ tok_list, const float* __restrict__ wgt_list,
             float* __restrict__ out, int n) {
    const int e    = blockIdx.y;
    const int tile = blockIdx.x;
    const int cnt  = counts[e];
    if (tile * MTILE >= cnt) return;

    __shared__ __align__(16) char smem[MTILE * HDIM * 2];  // 64 KB shared region
    float*  xF   = (float*)smem;               // 32 KB f32 staged x (temporary)
    __bf16* actA = (__bf16*)smem;              // 64 KB bf16 act [m][h] (later)
    __shared__ __bf16 xA[MTILE * CDIM];        // 16 KB bf16 x tile [m][k]
    __shared__ int    s_tok[MTILE];
    __shared__ float  s_wgt[MTILE];

    const int tid = threadIdx.x;
    if (tid < MTILE) {
        int idx = tile * MTILE + tid;
        int tok = (idx < cnt) ? tok_list[(size_t)e * n + idx] : -1;
        s_tok[tid] = tok;
        s_wgt[tid] = (idx < cnt) ? wgt_list[(size_t)e * n + idx] : 0.f;
    }
    __syncthreads();

    // ---- async DMA: 32 rows x 1KB = 2048 x 16B chunks, 8 per thread ----
    {
        const unsigned long long xbase = (unsigned long long)(uintptr_t)x;
        const unsigned ldsbase = (unsigned)(uintptr_t)xF;
        #pragma unroll
        for (int it = 0; it < 8; ++it) {
            int chunk   = it * 256 + tid;
            int byteoff = chunk << 4;
            int row     = byteoff >> 10;          // 1024 B per row
            int col     = byteoff & 1023;
            int tok     = s_tok[row];
            unsigned goff  = (unsigned)((tok >= 0 ? tok : 0) * (CDIM * 4) + col);
            unsigned laddr = ldsbase + (unsigned)byteoff;
            asm volatile("global_load_async_to_lds_b128 %0, %1, %2"
                         :: "v"(laddr), "v"(goff), "s"(xbase) : "memory");
        }
        asm volatile("s_wait_asynccnt 0" ::: "memory");
    }
    __syncthreads();

    // ---- convert staged tile to bf16 (zero padding rows) ----
    for (int i = tid; i < MTILE * CDIM; i += 256) {
        int m = i >> 8;
        float v = (s_tok[m] >= 0) ? xF[i] : 0.f;
        xA[i] = (__bf16)v;
    }
    __syncthreads();   // xF dead from here; its bytes become actA

    const int wave = tid >> 5;
    const int lane = tid & 31;
    const int mrow = lane & 15;   // A: row; B/C/D: column
    const int half = lane >> 4;

    const float* w1e = W1 + (size_t)e * CDIM * HDIM;
    const float* w2e = W2 + (size_t)e * HDIM * CDIM;

    // ---- GEMM1: two passes of 4 N-tiles; per kc: 2 A frags, 4 B frags,
    //      8 WMMAs; 64 accumulator VGPRs per pass keeps loads pipelined ----
    #pragma unroll 1
    for (int np = 0; np < 2; ++np) {
        v8f acc1[2][4];
        #pragma unroll
        for (int mt = 0; mt < 2; ++mt)
            #pragma unroll
            for (int i = 0; i < 4; ++i) acc1[mt][i] = (v8f){};

        for (int kc = 0; kc < CDIM / 32; ++kc) {
            const int k0 = kc * 32;
            v16bf a[2];
            #pragma unroll
            for (int mt = 0; mt < 2; ++mt) {
                U8x32 at;
                const __bf16* arow = xA + (mt * 16 + mrow) * CDIM + k0;
                at.a = *(const uint4*)(arow + half * 16);
                at.b = *(const uint4*)(arow + half * 16 + 8);
                a[mt] = __builtin_bit_cast(v16bf, at);
            }
            if (kc + 1 < CDIM / 32)
                __builtin_prefetch(&w1e[(size_t)(k0 + 32 + half * 16) * HDIM +
                                        wave * 16 + mrow], 0, 1);
            #pragma unroll
            for (int i = 0; i < 4; ++i) {
                const int nt = wave + (np * 4 + i) * 8;
                float tf[16];                   // load phase (clause-friendly)
                #pragma unroll
                for (int j = 0; j < 16; ++j)
                    tf[j] = w1e[(size_t)(k0 + half * 16 + j) * HDIM +
                                nt * 16 + mrow];
                v16bf b;                        // convert phase
                #pragma unroll
                for (int j = 0; j < 16; ++j) b[j] = (__bf16)tf[j];
                acc1[0][i] = __builtin_amdgcn_wmma_f32_16x16x32_bf16(
                                 false, a[0], false, b, (short)0, acc1[0][i],
                                 false, false);
                acc1[1][i] = __builtin_amdgcn_wmma_f32_16x16x32_bf16(
                                 false, a[1], false, b, (short)0, acc1[1][i],
                                 false, false);
            }
        }
        // relu^2 -> bf16 LDS (D layout: col = mrow, row = mt*16 + r + half*8)
        #pragma unroll
        for (int mt = 0; mt < 2; ++mt)
            #pragma unroll
            for (int i = 0; i < 4; ++i) {
                const int nt = wave + (np * 4 + i) * 8;
                #pragma unroll
                for (int r = 0; r < 8; ++r) {
                    float v = fmaxf(acc1[mt][i][r], 0.f);
                    actA[(mt * 16 + r + half * 8) * HDIM + nt * 16 + mrow] =
                        (__bf16)(v * v);
                }
            }
    }
    __syncthreads();

    // ---- GEMM2: per kc: 2 A frags, 2 B frags, 4 WMMAs (32 acc VGPRs) ----
    const int nt0 = wave, nt1 = wave + 8;
    v8f acc2[2][2];
    #pragma unroll
    for (int mt = 0; mt < 2; ++mt)
        #pragma unroll
        for (int j = 0; j < 2; ++j) acc2[mt][j] = (v8f){};

    for (int kc = 0; kc < HDIM / 32; ++kc) {
        const int k0 = kc * 32;
        v16bf a[2];
        #pragma unroll
        for (int mt = 0; mt < 2; ++mt) {
            U8x32 at;
            const __bf16* arow = actA + (mt * 16 + mrow) * HDIM + k0;
            at.a = *(const uint4*)(arow + half * 16);
            at.b = *(const uint4*)(arow + half * 16 + 8);
            a[mt] = __builtin_bit_cast(v16bf, at);
        }
        if (kc + 1 < HDIM / 32)
            __builtin_prefetch(&w2e[(size_t)(k0 + 32 + half * 16) * CDIM +
                                    nt0 * 16 + mrow], 0, 1);
        float t0[16], t1[16];
        #pragma unroll
        for (int j = 0; j < 16; ++j) {
            const size_t rowoff = (size_t)(k0 + half * 16 + j) * CDIM + mrow;
            t0[j] = w2e[rowoff + nt0 * 16];
            t1[j] = w2e[rowoff + nt1 * 16];
        }
        v16bf b0, b1;
        #pragma unroll
        for (int j = 0; j < 16; ++j) { b0[j] = (__bf16)t0[j]; b1[j] = (__bf16)t1[j]; }
        acc2[0][0] = __builtin_amdgcn_wmma_f32_16x16x32_bf16(
                         false, a[0], false, b0, (short)0, acc2[0][0], false, false);
        acc2[1][0] = __builtin_amdgcn_wmma_f32_16x16x32_bf16(
                         false, a[1], false, b0, (short)0, acc2[1][0], false, false);
        acc2[0][1] = __builtin_amdgcn_wmma_f32_16x16x32_bf16(
                         false, a[0], false, b1, (short)0, acc2[0][1], false, false);
        acc2[1][1] = __builtin_amdgcn_wmma_f32_16x16x32_bf16(
                         false, a[1], false, b1, (short)0, acc2[1][1], false, false);
    }
    #pragma unroll
    for (int mt = 0; mt < 2; ++mt)
        #pragma unroll
        for (int r = 0; r < 8; ++r) {
            const int mm = mt * 16 + r + half * 8;
            const int tok = s_tok[mm];
            if (tok >= 0) {
                const float w = s_wgt[mm];
                atomicAdd(&out[(size_t)tok * CDIM + nt0 * 16 + mrow],
                          acc2[mt][0][r] * w);
                atomicAdd(&out[(size_t)tok * CDIM + nt1 * 16 + mrow],
                          acc2[mt][1][r] * w);
            }
        }
}

// ---------------------------------------------------------------------------
extern "C" void kernel_launch(void* const* d_in, const int* in_sizes, int n_in,
                              void* d_out, int out_size, void* d_ws, size_t ws_size,
                              hipStream_t stream) {
    const float* x  = (const float*)d_in[0];   // [n, 256]
    const float* Wg = (const float*)d_in[1];   // [256, 32]
    const float* W1 = (const float*)d_in[2];   // [32, 256, 1024]
    const float* W2 = (const float*)d_in[3];   // [32, 1024, 256]
    const int n = in_sizes[0] / CDIM;          // 2048

    float* outp = (float*)d_out;                       // [n, 256]
    float* wsp  = outp + (size_t)n * CDIM;             // [n, 32] (tuple output #2)

    char*  ws       = (char*)d_ws;
    int*   counts   = (int*)ws;                                       // E ints
    int*   top_i    = (int*)(ws + 256);                               // n*2
    float* top_w    = (float*)(ws + 256 + (size_t)n * 2 * sizeof(int));
    int*   tok_list = (int*)(ws + 256 + (size_t)n * 4 * sizeof(int)); // E*n
    float* wgt_list = (float*)((char*)tok_list + (size_t)E_EXP * n * sizeof(int));

    hipMemsetAsync(outp, 0, (size_t)n * CDIM * sizeof(float), stream);

    moe_gate<<<(n + 7) / 8, 256, 0, stream>>>(x, Wg, wsp, top_i, top_w, n);
    moe_compact<<<E_EXP, 32, 0, stream>>>(top_i, top_w, counts, tok_list, wgt_list, n);

    dim3 grid((n + MTILE - 1) / MTILE, E_EXP);  // worst-case tiles x experts
    moe_mlp<<<grid, 256, 0, stream>>>(x, W1, W2, counts, tok_list, wgt_list, outp, n);
}
